// StyleLossPass1_23081154249115
// MI455X (gfx1250) — matrix operand order repaired and checked by reading, not compiled, not run on hardware
//
#include <hip/hip_runtime.h>
#include <hip/hip_bf16.h>

#define HWQ 4096          // H*W
#define CCH 256           // channels
#define KP  2304          // C * 3*3 patch length
#define EPSV 1e-9f

typedef __attribute__((ext_vector_type(16))) __bf16 v16bf;
typedef __attribute__((ext_vector_type(8)))  __bf16 v8bf;
typedef __attribute__((ext_vector_type(8)))  float  v8f;
typedef __attribute__((ext_vector_type(2)))  float  v2f;

// --- TDM availability -------------------------------------------------------
#if defined(__AMDGCN__) && __has_builtin(__builtin_amdgcn_tensor_load_to_lds) && \
    __has_builtin(__builtin_amdgcn_s_wait_tensorcnt)
#define USE_TDM 1
#else
#define USE_TDM 0
#endif

#if __has_include(<hip/amd_detail/amd_gfx1250_TDM.h>)
#define TDM_6ARG 1
#else
#define TDM_6ARG 0
#endif

#define TILE_ELEMS (128 * 40)          // padded 128x32 bf16 tile in LDS
#define TILE_BYTES (TILE_ELEMS * 2)

__device__ __forceinline__ unsigned sortable_f32(float f) {
  unsigned u = __float_as_uint(f);
  return (u & 0x80000000u) ? ~u : (u | 0x80000000u);
}

#if USE_TDM
typedef __attribute__((ext_vector_type(4))) unsigned int v4u;
typedef __attribute__((ext_vector_type(8))) int v8i;
typedef __attribute__((ext_vector_type(4))) int v4i;

// Generic pointers to LDS carry the byte offset in the low 32 bits
// (flat aperture in [63:32], offset in [31:0]).
__device__ __forceinline__ unsigned lds_off(const void* p) {
  return (unsigned)(unsigned long long)(uintptr_t)p;
}

// D# for a 128-row x 32-col bf16 tile of a (4096 x KP) row-major tensor.
// LDS destination padded: 16 DWORDs (one 64B row) then 4 DWORDs pad -> stride
// 40 bf16, matching the compute-side layout.
__device__ __forceinline__ void tdm_load_tile(unsigned lds_addr,
                                              unsigned long long gaddr) {
  v4u g0;
  g0[0] = 1u;                                             // count=1 (valid), user mode
  g0[1] = lds_addr;                                       // LDS byte address
  g0[2] = (unsigned)(gaddr & 0xFFFFFFFFull);              // global addr [31:0]
  g0[3] = (unsigned)((gaddr >> 32) & 0x01FFFFFFull)       // global addr [56:32]
          | (2u << 30);                                   // type=2 ("image")
  v8i g1;
  g1[0] = (1 << 16)                                       // data_size = 2 bytes
          | (1 << 20)                                     // pad_enable
          | (3 << 22)                                     // pad_interval: 16 DWORDs
          | (3 << 25);                                    // pad_amount:   4 DWORDs
  g1[1] = (int)((KP & 0xFFFF) << 16);                     // tensor_dim0 = KP (lo16)
  g1[2] = (int)((KP >> 16) | ((HWQ & 0xFFFF) << 16));     // dim0 hi | dim1 lo
  g1[3] = (int)((HWQ >> 16) | (32 << 16));                // dim1 hi | tile_dim0 = 32
  g1[4] = 128;                                            // tile_dim1 = 128, tile_dim2 = 0
  g1[5] = KP;                                             // tensor_dim0_stride lo32
  g1[6] = 0;                                              // stride0 hi | stride1 lo
  g1[7] = 0;
  v4i gz = {0, 0, 0, 0};                                  // 2D tensor: groups 2/3 unused
#if TDM_6ARG
  v8i gz8 = {0, 0, 0, 0, 0, 0, 0, 0};
  __builtin_amdgcn_tensor_load_to_lds(g0, g1, gz, gz, gz8, 0);
#else
  __builtin_amdgcn_tensor_load_to_lds(g0, g1, gz, gz, 0);
#endif
}
#endif  // USE_TDM

// ---------------------------------------------------------------------------
// Build bf16 patch matrix P[pos][k], k = j*256 + c, j = (dy+1)*3 + (dx+1),
// reflect padding (jnp.pad 'reflect': -1 -> 1, 64 -> 62).
// grid = 4096 (positions), block = 256 (channels)
// ---------------------------------------------------------------------------
__global__ void __launch_bounds__(256)
build_patches_kernel(const float* __restrict__ fm, __bf16* __restrict__ P) {
  const int pos = blockIdx.x;
  const int c = threadIdx.x;
  const int y = pos >> 6, x = pos & 63;
#pragma unroll
  for (int j = 0; j < 9; ++j) {
    const int dy = j / 3 - 1, dx = j % 3 - 1;
    int yy = y + dy; yy = yy < 0 ? -yy : (yy > 63 ? 126 - yy : yy);
    int xx = x + dx; xx = xx < 0 ? -xx : (xx > 63 ? 126 - xx : xx);
    const float v = fm[(c << 12) + (yy << 6) + xx];
    P[(size_t)pos * KP + j * 256 + c] = (__bf16)v;
  }
}

// ---------------------------------------------------------------------------
// Patch L2 norms from the bf16 patch matrix. grid = 4096, block = 256.
// ---------------------------------------------------------------------------
__global__ void __launch_bounds__(256)
patch_norms_kernel(const __bf16* __restrict__ P, float* __restrict__ nrm) {
  __shared__ float red[256];
  const int pos = blockIdx.x, tid = threadIdx.x;
  float s = 0.f;
  for (int k = tid; k < KP; k += 256) {
    const float v = (float)P[(size_t)pos * KP + k];
    s += v * v;
  }
  red[tid] = s; __syncthreads();
  for (int w = 128; w > 0; w >>= 1) {
    if (tid < w) red[tid] += red[tid + w];
    __syncthreads();
  }
  if (tid == 0) nrm[pos] = sqrtf(red[0]);
}

// ---------------------------------------------------------------------------
// Fused normalized-score GEMM + argmax.
//   score[m][n] = <cp[m], sp[n]> / (cn[m]*sn[n] + EPS), argmax over n.
// grid = 256 blocks: m_block = bid&31 (128 rows), nsplit = bid>>5 (512 cols).
// block = 256 threads = 8 waves arranged 2(m) x 4(n); wave tile 64x32.
// Tile staging: double-buffered TDM (tensor_load_to_lds + s_wait_tensorcnt)
// when available; synchronous global->LDS copy otherwise.
// Cross-wave/block argmax via packed u64 atomicMax (LDS then global).
// ---------------------------------------------------------------------------
__global__ void __launch_bounds__(256)
score_argmax_kernel(const __bf16* __restrict__ cp, const __bf16* __restrict__ sp,
                    const float* __restrict__ cn, const float* __restrict__ sn,
                    unsigned long long* __restrict__ gbest) {
  __shared__ __bf16 As[2 * TILE_ELEMS];   // two 128x32 tiles, row stride 40 bf16
  __shared__ __bf16 Bs[2 * TILE_ELEMS];
  __shared__ float cnS[128];
  __shared__ float snS[128];
  __shared__ unsigned long long bestS[128];

  const int tid = threadIdx.x;
  const int lane = tid & 31;
  const int wid = tid >> 5;
  const int wave_m = wid >> 2;   // 0..1 -> 64 rows each
  const int wave_n = wid & 3;    // 0..3 -> 32 cols each
  const int m0 = (blockIdx.x & 31) * 128;
  const int nsplit = blockIdx.x >> 5;

  if (tid < 128) { cnS[tid] = cn[m0 + tid]; bestS[tid] = 0ull; }

  const int lrow = lane & 15;
  const int half = lane >> 4;
  const int khalfA = half * 8;   // A frag: K {kh..kh+7, kh+16..kh+23}
  const int kbaseB = half * 8;   // B frag: K {kb..kb+7, kb+16..kb+23}

  for (int nt = 0; nt < 4; ++nt) {
    const int n0 = nsplit * 512 + nt * 128;
    __syncthreads();
    if (tid < 128) snS[tid] = sn[n0 + tid];

    v8f acc[4][2] = {};

#if USE_TDM
    const int NCHUNK = KP / 32;
    const unsigned long long gaBase =
        (unsigned long long)(uintptr_t)cp + (unsigned long long)m0 * (KP * 2);
    const unsigned long long gbBase =
        (unsigned long long)(uintptr_t)sp + (unsigned long long)n0 * (KP * 2);
    const unsigned la = lds_off(As);
    const unsigned lb = lds_off(Bs);
    if (wid == 0) {                       // wave-level TDM issue (EXEC ignored)
      tdm_load_tile(la, gaBase);          // chunk 0 -> buffer 0
      tdm_load_tile(lb, gbBase);
    }
    for (int kc = 0; kc < NCHUNK; ++kc) {
      const int cur = kc & 1;
      if (wid == 0) {
        if (kc + 1 < NCHUNK) {            // prefetch chunk kc+1 into other buffer
          const unsigned nb = (unsigned)((kc + 1) & 1) * TILE_BYTES;
          const unsigned long long koff = (unsigned long long)(kc + 1) * 64ull;
          tdm_load_tile(la + nb, gaBase + koff);
          tdm_load_tile(lb + nb, gbBase + koff);
          __builtin_amdgcn_s_wait_tensorcnt(2);  // chunk kc resident
        } else {
          __builtin_amdgcn_s_wait_tensorcnt(0);
        }
      }
      __syncthreads();
      const __bf16* AsP = As + cur * TILE_ELEMS;
      const __bf16* BsP = Bs + cur * TILE_ELEMS;
#else
    for (int k = 0; k < KP; k += 32) {
      __syncthreads();
#pragma unroll
      for (int h = 0; h < 2; ++h) {
        const int chunk = tid + h * 256;      // 0..511
        const int row = chunk >> 2;
        const int kc8 = (chunk & 3) * 8;
        *(uint4*)&As[row * 40 + kc8] =
            *(const uint4*)(cp + (size_t)(m0 + row) * KP + k + kc8);
        *(uint4*)&Bs[row * 40 + kc8] =
            *(const uint4*)(sp + (size_t)(n0 + row) * KP + k + kc8);
      }
      if (k + 32 < KP) {
        __builtin_prefetch(cp + (size_t)(m0 + (tid >> 1)) * KP + k + 32, 0, 1);
        __builtin_prefetch(sp + (size_t)(n0 + (tid >> 1)) * KP + k + 32, 0, 1);
      }
      __syncthreads();
      const __bf16* AsP = As;
      const __bf16* BsP = Bs;
#endif

      v16bf bfrag[2];
#pragma unroll
      for (int ni = 0; ni < 2; ++ni) {
        const __bf16* bp = &BsP[(wave_n * 32 + ni * 16 + lrow) * 40 + kbaseB];
        const v8bf lo = *(const v8bf*)bp;
        const v8bf hi = *(const v8bf*)(bp + 16);
#pragma unroll
        for (int e = 0; e < 8; ++e) { bfrag[ni][e] = lo[e]; bfrag[ni][e + 8] = hi[e]; }
      }
#pragma unroll
      for (int mi = 0; mi < 4; ++mi) {
        const __bf16* ap = &AsP[(wave_m * 64 + mi * 16 + lrow) * 40 + khalfA];
        const v8bf lo = *(const v8bf*)ap;
        const v8bf hi = *(const v8bf*)(ap + 16);
        v16bf afrag;
#pragma unroll
        for (int e = 0; e < 8; ++e) { afrag[e] = lo[e]; afrag[e + 8] = hi[e]; }
#pragma unroll
        for (int ni = 0; ni < 2; ++ni) {
          acc[mi][ni] = __builtin_amdgcn_wmma_f32_16x16x32_bf16(
              false, afrag, false, bfrag[ni], (short)0, acc[mi][ni], false, false);
        }
      }
#if USE_TDM
      __syncthreads();   // all waves done reading buf[cur] before it is re-filled
#endif
    }
    __syncthreads();

    // Epilogue: normalize + argmax. C layout: lane l, VGPR i ->
    // row = i + 8*(l>>4), col = l&15 within each 16x16 tile.
#pragma unroll
    for (int mi = 0; mi < 4; ++mi) {
#pragma unroll
      for (int i = 0; i < 8; ++i) {
        const int row = wave_m * 64 + mi * 16 + half * 8 + i;
        const float cnv = cnS[row];
        const int c0 = wave_n * 32 + lrow;
        const int c1 = c0 + 16;
        const float v0 = acc[mi][0][i] / (cnv * snS[c0] + EPSV);
        const float v1 = acc[mi][1][i] / (cnv * snS[c1] + EPSV);
        float v; int nidx;
        if (v1 > v0) { v = v1; nidx = n0 + c1; } else { v = v0; nidx = n0 + c0; }
        // argmax across the 16-lane half (xor 1,2,4,8 stays within the half)
#pragma unroll
        for (int mm = 8; mm >= 1; mm >>= 1) {
          const float ov = __shfl_xor(v, mm, 32);
          const int oi = __shfl_xor(nidx, mm, 32);
          if (ov > v) { v = ov; nidx = oi; }
        }
        if (lrow == 0) {
          const unsigned long long pk =
              ((unsigned long long)sortable_f32(v) << 32) | (unsigned)nidx;
          atomicMax(&bestS[row], pk);
        }
      }
    }
  }
  __syncthreads();
  if (tid < 128) atomicMax(&gbest[m0 + tid], bestS[tid]);
}

// ---------------------------------------------------------------------------
__global__ void init_kernel(unsigned long long* __restrict__ gbest) {
  const int i = blockIdx.x * 256 + threadIdx.x;
  if (i < HWQ) gbest[i] = 0ull;
}

__global__ void unpack_idx_kernel(const unsigned long long* __restrict__ gbest,
                                  int* __restrict__ idx) {
  const int i = blockIdx.x * 256 + threadIdx.x;
  if (i < HWQ) idx[i] = (int)(gbest[i] & 0xFFFFFFFFull);
}

// mask_exp sum over (1,256,64,64) = 256 * sum(mask)
__global__ void __launch_bounds__(256)
mask_sum_kernel(const float* __restrict__ mask, float* __restrict__ msum) {
  __shared__ float red[256];
  float s = 0.f;
  for (int i = threadIdx.x; i < HWQ; i += 256) s += mask[i];
  red[threadIdx.x] = s; __syncthreads();
  for (int w = 128; w > 0; w >>= 1) {
    if (threadIdx.x < w) red[threadIdx.x] += red[threadIdx.x + w];
    __syncthreads();
  }
  if (threadIdx.x == 0) msum[0] = red[0] * (float)CCH;
}

// A1 = corr * mask (gathered style), A2 = input * mask. grid = 256 (channel).
__global__ void __launch_bounds__(256)
gather_mask_kernel(const float* __restrict__ style, const float* __restrict__ input,
                   const float* __restrict__ mask, const int* __restrict__ idx,
                   float* __restrict__ A1, float* __restrict__ A2,
                   __bf16* __restrict__ A1b, __bf16* __restrict__ A2b) {
  const int c = blockIdx.x;
  for (int x = threadIdx.x; x < HWQ; x += 256) {
    const int j = idx[x];
    const float mv = mask[x];
    const float a1 = style[(size_t)c * HWQ + j] * mv;
    const float a2 = input[(size_t)c * HWQ + x] * mv;
    A1[(size_t)c * HWQ + x] = a1;
    A2[(size_t)c * HWQ + x] = a2;
    A1b[(size_t)c * HWQ + x] = (__bf16)a1;
    A2b[(size_t)c * HWQ + x] = (__bf16)a2;
  }
}

// ---------------------------------------------------------------------------
// Gram: G = A @ A^T, A is 256x4096. One wave per 16x16 output tile.
// grid = 32, block = 256 (8 waves) -> 256 tiles.
// Preferred path: full-f32 WMMA (16x16x4). Fallback: bf16 WMMA.
// ---------------------------------------------------------------------------
__global__ void __launch_bounds__(256)
gram_kernel(const float* __restrict__ A, const __bf16* __restrict__ Ab,
            float* __restrict__ G) {
  const int tid = threadIdx.x;
  const int lane = tid & 31;
  const int wid = tid >> 5;
  const int tile = blockIdx.x * 8 + wid;  // 0..255
  const int tm = tile >> 4, tn = tile & 15;
  const int lrow = lane & 15;
  const int half = lane >> 4;
  v8f acc = {};
#if __has_builtin(__builtin_amdgcn_wmma_f32_16x16x4_f32)
  // f32 A-frag (16x4): lane holds row lane&15, K = {2*half, 2*half+1}
  const float* arow = A + (size_t)(tm * 16 + lrow) * HWQ;
  const float* brow = A + (size_t)(tn * 16 + lrow) * HWQ;
  const int lk = half * 2;
  for (int k = 0; k < HWQ; k += 4) {
    const v2f a = *(const v2f*)(arow + k + lk);
    const v2f b = *(const v2f*)(brow + k + lk);
    acc = __builtin_amdgcn_wmma_f32_16x16x4_f32(false, a, false, b, (short)0,
                                                acc, false, false);
  }
  (void)Ab;
#else
  const __bf16* arow = Ab + (size_t)(tm * 16 + lrow) * HWQ;
  const __bf16* brow = Ab + (size_t)(tn * 16 + lrow) * HWQ;
  const int ka = half * 8;
  const int kb = half * 16;
  for (int k = 0; k < HWQ; k += 32) {
    const v8bf alo = *(const v8bf*)(arow + k + ka);
    const v8bf ahi = *(const v8bf*)(arow + k + ka + 16);
    const v8bf blo = *(const v8bf*)(brow + k + kb);
    const v8bf bhi = *(const v8bf*)(brow + k + kb + 8);
    v16bf af, bfv;
#pragma unroll
    for (int e = 0; e < 8; ++e) {
      af[e] = alo[e]; af[e + 8] = ahi[e];
      bfv[e] = blo[e]; bfv[e + 8] = bhi[e];
    }
    acc = __builtin_amdgcn_wmma_f32_16x16x32_bf16(false, af, false, bfv, (short)0,
                                                  acc, false, false);
  }
  (void)A;
#endif
  const int col = tn * 16 + lrow;
#pragma unroll
  for (int i = 0; i < 8; ++i) {
    const int row = tm * 16 + i + half * 8;
    G[row * CCH + col] = acc[i];
  }
}

// loss = mean(((G_in - G_corr)/msum)^2) * 100
__global__ void __launch_bounds__(256)
loss_kernel(const float* __restrict__ G1, const float* __restrict__ G2,
            const float* __restrict__ msum, float* __restrict__ out) {
  __shared__ float red[256];
  const float inv = 1.0f / msum[0];
  float s = 0.f;
  for (int i = threadIdx.x; i < CCH * CCH; i += 256) {
    const float d = (G1[i] - G2[i]) * inv;
    s += d * d;
  }
  red[threadIdx.x] = s; __syncthreads();
  for (int w = 128; w > 0; w >>= 1) {
    if (threadIdx.x < w) red[threadIdx.x] += red[threadIdx.x + w];
    __syncthreads();
  }
  if (threadIdx.x == 0) out[0] = red[0] * (100.0f / (float)(CCH * CCH));
}

// ---------------------------------------------------------------------------
extern "C" void kernel_launch(void* const* d_in, const int* in_sizes, int n_in,
                              void* d_out, int out_size, void* d_ws, size_t ws_size,
                              hipStream_t stream) {
  const float* content = (const float*)d_in[0];
  const float* style   = (const float*)d_in[1];
  const float* input   = (const float*)d_in[2];
  const float* mask    = (const float*)d_in[3];
  float* out = (float*)d_out;
  (void)in_sizes; (void)n_in; (void)out_size; (void)ws_size;

  char* w = (char*)d_ws;
  size_t o = 0;
  auto take = [&](size_t b) -> void* {
    o = (o + 255) & ~(size_t)255;
    void* p = w + o;
    o += b;
    return p;
  };
  __bf16* cp = (__bf16*)take((size_t)HWQ * KP * 2);          // 18.9 MB
  __bf16* sp = (__bf16*)take((size_t)HWQ * KP * 2);          // 18.9 MB
  float* cn = (float*)take((size_t)HWQ * 4);
  float* sn = (float*)take((size_t)HWQ * 4);
  unsigned long long* gbest = (unsigned long long*)take((size_t)HWQ * 8);
  int* idx = (int*)take((size_t)HWQ * 4);
  float* A1 = (float*)take((size_t)CCH * HWQ * 4);           // corr*mask
  float* A2 = (float*)take((size_t)CCH * HWQ * 4);           // input*mask
  __bf16* A1b = (__bf16*)take((size_t)CCH * HWQ * 2);
  __bf16* A2b = (__bf16*)take((size_t)CCH * HWQ * 2);
  float* G1 = (float*)take((size_t)CCH * CCH * 4);           // gram(input*mask)
  float* G2 = (float*)take((size_t)CCH * CCH * 4);           // gram(corr*mask)
  float* msum = (float*)take(256);

  hipLaunchKernelGGL(init_kernel, dim3(16), dim3(256), 0, stream, gbest);
  hipLaunchKernelGGL(mask_sum_kernel, dim3(1), dim3(256), 0, stream, mask, msum);
  hipLaunchKernelGGL(build_patches_kernel, dim3(HWQ), dim3(CCH), 0, stream, content, cp);
  hipLaunchKernelGGL(build_patches_kernel, dim3(HWQ), dim3(CCH), 0, stream, style, sp);
  hipLaunchKernelGGL(patch_norms_kernel, dim3(HWQ), dim3(256), 0, stream, cp, cn);
  hipLaunchKernelGGL(patch_norms_kernel, dim3(HWQ), dim3(256), 0, stream, sp, sn);
  hipLaunchKernelGGL(score_argmax_kernel, dim3(256), dim3(256), 0, stream,
                     cp, sp, cn, sn, gbest);
  hipLaunchKernelGGL(unpack_idx_kernel, dim3(16), dim3(256), 0, stream, gbest, idx);
  hipLaunchKernelGGL(gather_mask_kernel, dim3(CCH), dim3(256), 0, stream,
                     style, input, mask, idx, A1, A2, A1b, A2b);
  hipLaunchKernelGGL(gram_kernel, dim3(32), dim3(256), 0, stream, A2, A2b, G1);
  hipLaunchKernelGGL(gram_kernel, dim3(32), dim3(256), 0, stream, A1, A1b, G2);
  hipLaunchKernelGGL(loss_kernel, dim3(1), dim3(256), 0, stream, G1, G2, msum, out);
}
